// OutputLayer_62096637166424
// MI455X (gfx1250) — matrix operand order repaired and checked by reading, not compiled, not run on hardware
//
#include <hip/hip_runtime.h>
#include <hip/hip_bf16.h>
#include <cstdint>

// ---------------- problem constants ----------------
#define B_  8
#define T_  2000
#define D_  512
#define Y_  8921
#define NT_ 125          // T/16 chunks of 16 timesteps
#define KC_ 16           // D/32 K-chunks per WMMA
#define YP_ 9216         // Y padded to multiple of 64
#define YTILES_ 576      // YP/16
#define WAVES_ 4         // waves per block (wave32) -> 64 y rows per block
#define THREADS_ (WAVES_ * 32)
#define CHUNK_ELEMS_ (KC_ * 32 * 16)   // 8192 bf16 = 16 KB per T-chunk

typedef __attribute__((ext_vector_type(16))) __bf16 v16bf;
typedef __attribute__((ext_vector_type(8)))  float  v8f;

// ws layout (bytes)
#define XB_BYTES  ((size_t)B_ * NT_ * CHUNK_ELEMS_ * 2)    // 16,384,000
#define WB_BYTES  ((size_t)YTILES_ * KC_ * 32 * 16 * 2)    //  9,437,184

// ---------------------------------------------------------------------------
// CDNA5 async global->LDS support (device-side only; launch config identical
// on both paths so host/device stay consistent).
// Builtin signature (from hipcc diagnostic): param 1 is
//   'int __vector(4) __device__*'  == addrspace(1) pointer to int4.
// ---------------------------------------------------------------------------
#if defined(__AMDGCN__) && __has_builtin(__builtin_amdgcn_global_load_async_to_lds_b128)
#define HAVE_ASYNC_LDS 1
typedef int v4i_vs __attribute__((vector_size(16)));
typedef __attribute__((address_space(1))) v4i_vs gv4i;   // global int4
typedef __attribute__((address_space(3))) v4i_vs lv4i;   // LDS int4
__device__ __forceinline__ void async_ld_b128(const void* g, void* l) {
    // AS1 pointer: same 64-bit value as generic global pointer.
    // AS3 pointer: low 32 bits of the flat LDS address (aperture in high bits).
    __builtin_amdgcn_global_load_async_to_lds_b128(
        (gv4i*)(uintptr_t)g,
        (lv4i*)(uint32_t)(uintptr_t)l,
        0, 0);
}
#if __has_builtin(__builtin_amdgcn_s_wait_asynccnt)
#define WAIT_ASYNC(n) __builtin_amdgcn_s_wait_asynccnt(n)
#else
#define WAIT_ASYNC(n) asm volatile("s_wait_asynccnt %0" :: "i"(n) : "memory")
#endif
#else
#define HAVE_ASYNC_LDS 0
#endif

// ---------------------------------------------------------------------------
// Reorder x (B,T,D) fp32 -> bf16 in WMMA B-fragment order:
//   slot = (((b*NT + tc)*KC + kc)*32 + lane), 16 bf16 per slot.
//   B-matrix (32x16, KxN): N = lane%16 (=t within chunk),
//   K = kc*32 + (lane<16 ? 0 : 8) + j + (j<8 ? 0 : 8), j=0..15.
// ---------------------------------------------------------------------------
__global__ void reorder_x_kernel(const float* __restrict__ x,
                                 __bf16* __restrict__ xb) {
    int gid = blockIdx.x * 256 + threadIdx.x;           // slot id
    if (gid >= B_ * NT_ * KC_ * 32) return;
    int lane = gid & 31;
    int kc   = (gid >> 5) & 15;
    int tc   = (gid >> 9) % NT_;
    int b    = gid / (NT_ * KC_ * 32);
    int t    = tc * 16 + (lane & 15);
    int kb   = kc * 32 + ((lane & 16) ? 8 : 0);
    const float* src = x + ((size_t)(b * T_ + t)) * D_ + kb;
    __bf16* dst = xb + (size_t)gid * 16;
#pragma unroll
    for (int j = 0; j < 16; ++j) dst[j] = (__bf16)src[j + (j < 8 ? 0 : 8)];
}

// ---------------------------------------------------------------------------
// Reorder weights (Y,D) fp32 -> bf16 in WMMA A-fragment order, zero-pad Y->YP:
//   slot = ((ry*KC + kc)*32 + lane), 16 bf16 per slot.
//   A-matrix (16x32, MxK): M = lane%16 (=y within tile),
//   elems 0..7  : K = kc*32 + (lane<16 ? 0  : 8 ) + e
//   elems 8..15 : K = kc*32 + (lane<16 ? 16 : 24) + (e-8)
// ---------------------------------------------------------------------------
__global__ void reorder_w_kernel(const float* __restrict__ w,
                                 __bf16* __restrict__ wb) {
    int gid = blockIdx.x * 256 + threadIdx.x;
    if (gid >= YTILES_ * KC_ * 32) return;
    int lane = gid & 31;
    int kc   = (gid >> 5) & 15;
    int ry   = gid >> 9;
    int y    = ry * 16 + (lane & 15);
    int b0   = kc * 32 + ((lane & 16) ? 8 : 0);
    int b1   = kc * 32 + ((lane & 16) ? 24 : 16);
    __bf16* dst = wb + (size_t)gid * 16;
    if (y < Y_) {
        const float* src = w + (size_t)y * D_;
#pragma unroll
        for (int j = 0; j < 8; ++j) dst[j]     = (__bf16)src[b0 + j];
#pragma unroll
        for (int j = 0; j < 8; ++j) dst[8 + j] = (__bf16)src[b1 + j];
    } else {
#pragma unroll
        for (int j = 0; j < 16; ++j) dst[j] = (__bf16)0.0f;
    }
}

// ---------------------------------------------------------------------------
// Main fused kernel: per (y-tile, batch), stream T with online softmax.
//   S = U_tile * x_chunk^T   (16x16, via 16 WMMA over K=512)
//   V = F_tile * x_chunk^T   (16x16, via 16 WMMA over K=512)
// x chunks double-buffered in LDS, filled by async global->LDS when available.
// ---------------------------------------------------------------------------
__global__ __launch_bounds__(THREADS_, 1)
void attn_pool_kernel(const __bf16* __restrict__ xb,
                      const __bf16* __restrict__ Ub,
                      const __bf16* __restrict__ Fb,
                      const float* __restrict__ bias,
                      float* __restrict__ out) {
    __shared__ __attribute__((aligned(32))) __bf16 xlds[2 * CHUNK_ELEMS_]; // 32 KB

    const int tid  = threadIdx.x;
    const int wave = tid >> 5;
    const int lane = tid & 31;
    const int b    = blockIdx.y;
    const int ry   = blockIdx.x * WAVES_ + wave;         // 16-row y tile id

    // Per-wave weight fragments for full K=512, resident in VGPRs (256 VGPRs).
    v16bf ufrag[KC_], ffrag[KC_];
#pragma unroll
    for (int kc = 0; kc < KC_; ++kc) {
        size_t off = (((size_t)ry * KC_ + kc) * 32 + lane) * 16;
        ufrag[kc] = *(const v16bf*)(Ub + off);
        ffrag[kc] = *(const v16bf*)(Fb + off);
    }

    // Online-softmax state: 8 rows per lane (M = r + 8*(lane/16)).
    float mrow[8], lrow[8], arow[8];
#pragma unroll
    for (int r = 0; r < 8; ++r) { mrow[r] = -1.0e30f; lrow[r] = 0.0f; arow[r] = 0.0f; }

    const __bf16* xbatch = xb + (size_t)b * NT_ * CHUNK_ELEMS_;

#if HAVE_ASYNC_LDS
    // ---- async double-buffered pipeline: 8 x b128 per thread per chunk ----
    {
        const char* g = (const char*)xbatch + tid * 128;
        char*       l = (char*)xlds + tid * 128;
#pragma unroll
        for (int i = 0; i < 8; ++i) async_ld_b128(g + i * 16, l + i * 16);
    }
    for (int tc = 0; tc < NT_; ++tc) {
        const int cur = tc & 1;
        if (tc + 1 < NT_) {
            const char* g = (const char*)(xbatch + (size_t)(tc + 1) * CHUNK_ELEMS_) + tid * 128;
            char*       l = (char*)(xlds + (size_t)(1 - cur) * CHUNK_ELEMS_) + tid * 128;
#pragma unroll
            for (int i = 0; i < 8; ++i) async_ld_b128(g + i * 16, l + i * 16);
            WAIT_ASYNC(8);      // current chunk's 8 ops retired, next 8 in flight
        } else {
            WAIT_ASYNC(0);
        }
        __syncthreads();        // current buffer visible to all waves
        const __bf16* xcur = xlds + (size_t)cur * CHUNK_ELEMS_;
#else
    // ---- fallback: register-staged copy, single buffer ----
    uint4 pre[8];
    {
        const uint4* g = (const uint4*)xbatch + tid * 8;
#pragma unroll
        for (int i = 0; i < 8; ++i) pre[i] = g[i];
    }
    for (int tc = 0; tc < NT_; ++tc) {
        __syncthreads();                       // everyone done reading xlds
        {
            uint4* l = (uint4*)xlds + tid * 8;
#pragma unroll
            for (int i = 0; i < 8; ++i) l[i] = pre[i];
        }
        __syncthreads();                       // xlds ready
        if (tc + 1 < NT_) {                    // overlap next global fetch w/ math
            const uint4* g = (const uint4*)(xbatch + (size_t)(tc + 1) * CHUNK_ELEMS_) + tid * 8;
#pragma unroll
            for (int i = 0; i < 8; ++i) pre[i] = g[i];
        }
        const __bf16* xcur = xlds;
#endif

        v8f S = {}; v8f V = {};
#pragma unroll
        for (int kc = 0; kc < KC_; ++kc) {
            v16bf bfrag = *(const v16bf*)(xcur + ((size_t)kc * 32 + lane) * 16);
            S = __builtin_amdgcn_wmma_f32_16x16x32_bf16(false, ufrag[kc], false, bfrag,
                                                        (short)0, S, false, false);
            V = __builtin_amdgcn_wmma_f32_16x16x32_bf16(false, ffrag[kc], false, bfrag,
                                                        (short)0, V, false, false);
        }

        // Per-lane online softmax accumulation (one t per lane per row).
#pragma unroll
        for (int r = 0; r < 8; ++r) {
            float s  = S[r];
            float v  = V[r];
            float mn = fmaxf(mrow[r], s);
            float co = __expf(mrow[r] - mn);
            float e  = __expf(s - mn);
            lrow[r] = lrow[r] * co + e;
            arow[r] = arow[r] * co + e * v;
            mrow[r] = mn;
        }
#if HAVE_ASYNC_LDS
        __syncthreads();        // all waves done with current buffer
#endif
    }

    // Merge the 16 N-lane partials per row (log-sum-exp combine), then emit.
#pragma unroll
    for (int r = 0; r < 8; ++r) {
        float m = mrow[r], L = lrow[r], A = arow[r];
#pragma unroll
        for (int off = 8; off >= 1; off >>= 1) {
            float m2 = __shfl_xor(m, off, 16);
            float L2 = __shfl_xor(L, off, 16);
            float A2 = __shfl_xor(A, off, 16);
            float mn = fmaxf(m, m2);
            float c1 = __expf(m - mn);
            float c2 = __expf(m2 - mn);
            L = L * c1 + L2 * c2;
            A = A * c1 + A2 * c2;
            m = mn;
        }
        if ((lane & 15) == 0) {
            int y = ry * 16 + r + ((lane >> 4) << 3);
            if (y < Y_) out[(size_t)b * Y_ + y] = A / L + bias[y];
        }
    }
}

// ---------------------------------------------------------------------------
// BCE-with-logits mean over all (B,Y) logits -> single scalar.
// ---------------------------------------------------------------------------
__global__ void bce_loss_kernel(const float* __restrict__ logits,
                                const float* __restrict__ target,
                                float* __restrict__ loss_out) {
    __shared__ float red[256];
    float acc = 0.0f;
    for (int i = threadIdx.x; i < B_ * Y_; i += 256) {
        float z = logits[i];
        float t = target[i];
        acc += fmaxf(z, 0.0f) - z * t + log1pf(__expf(-fabsf(z)));
    }
    red[threadIdx.x] = acc;
    __syncthreads();
    for (int s = 128; s > 0; s >>= 1) {
        if (threadIdx.x < s) red[threadIdx.x] += red[threadIdx.x + s];
        __syncthreads();
    }
    if (threadIdx.x == 0) loss_out[0] = red[0] / (float)(B_ * Y_);
}

// ---------------------------------------------------------------------------
extern "C" void kernel_launch(void* const* d_in, const int* in_sizes, int n_in,
                              void* d_out, int out_size, void* d_ws, size_t ws_size,
                              hipStream_t stream) {
    const float* x      = (const float*)d_in[0];   // (B,T,D)
    const float* target = (const float*)d_in[1];   // (B,Y)
    // d_in[2] = text_inputs (unused)
    const float* U      = (const float*)d_in[3];   // (Y,D)
    const float* F      = (const float*)d_in[4];   // (Y,D)
    const float* bias   = (const float*)d_in[5];   // (Y,)
    float* out = (float*)d_out;                    // logits (B*Y) then loss (1)

    char* ws = (char*)d_ws;
    __bf16* xb = (__bf16*)ws;
    __bf16* Ub = (__bf16*)(ws + XB_BYTES);
    __bf16* Fb = (__bf16*)(ws + XB_BYTES + WB_BYTES);

    // One-time bf16 conversion + WMMA-fragment reordering.
    reorder_x_kernel<<<(B_ * NT_ * KC_ * 32 + 255) / 256, 256, 0, stream>>>(x, xb);
    reorder_w_kernel<<<(YTILES_ * KC_ * 32 + 255) / 256, 256, 0, stream>>>(U, Ub);
    reorder_w_kernel<<<(YTILES_ * KC_ * 32 + 255) / 256, 256, 0, stream>>>(F, Fb);

    // Fused attention-pool + logits.
    dim3 grid(YP_ / (16 * WAVES_), B_);            // (144, 8)
    attn_pool_kernel<<<grid, THREADS_, 0, stream>>>(xb, Ub, Fb, bias, out);

    // Scalar BCE loss.
    bce_loss_kernel<<<1, 256, 0, stream>>>(out, target, out + (size_t)B_ * Y_);
}
